// SuperGlue_7902739825096
// MI455X (gfx1250) — compile-verified, matrix operand order checked by reading, not run on hardware
//
#include <hip/hip_runtime.h>
#include <hip/hip_bf16.h>
#include <math.h>

typedef __attribute__((ext_vector_type(8)))  __bf16 v8bf;
typedef __attribute__((ext_vector_type(16))) __bf16 v16bf;
typedef __attribute__((ext_vector_type(8)))  float  v8f;

#define NTOK 1024
#define ZLD  1056                    // padded leading dim for 1025x1025 OT matrix
#define NORMC  (-7.624618986159398f) // -log(2048)
#define LOG1024 (6.931471805599453f)

union V16U { v16bf v; v8bf h[2]; };

__device__ __forceinline__ v16bf load_ab(const __bf16* base, int lo) {
  // Per ISA 16-bit A/B layout: lane<16 -> K[0..7],K[16..23]; lane>=16 -> K[8..15],K[24..31]
  V16U u;
  u.h[0] = *(const v8bf*)(base + lo);
  u.h[1] = *(const v8bf*)(base + lo + 16);
  return u.v;
}

// ---------------------------------------------------------------------------
// Generic WMMA GEMM:  Yt(n, M) = W(M,K) * Xt(n,K)^T + bias   (N tokens = 1024)
// One wave computes a 16(M) x 64(N) strip; 4 wmma per K-step with A reuse.
// ---------------------------------------------------------------------------
__global__ void gemm_wmma(const __bf16* __restrict__ W, const __bf16* __restrict__ Xt,
                          const float* __restrict__ bias, float* __restrict__ Yt,
                          int M, int K) {
  int wid  = (blockIdx.x * blockDim.x + threadIdx.x) >> 5;
  int lane = threadIdx.x & 31;
  int mTiles = M >> 4;
  int mt = wid % mTiles;
  int ns = wid / mTiles;
  int m0 = mt << 4, n0 = ns << 6;
  int l15 = lane & 15, half = lane >> 4;
  int lo = half << 3, off8 = half << 3;

  const __bf16* wrow = W + (size_t)(m0 + l15) * K;
  v8f binit = *(const v8f*)(bias + m0 + off8);
  v8f acc[4];
#pragma unroll
  for (int t = 0; t < 4; ++t) acc[t] = binit;

  for (int k = 0; k < K; k += 32) {
    v16bf a = load_ab(wrow + k, lo);
#pragma unroll
    for (int t = 0; t < 4; ++t) {
      const __bf16* xrow = Xt + (size_t)(n0 + t * 16 + l15) * K + k;
      v16bf b = load_ab(xrow, lo);
      acc[t] = __builtin_amdgcn_wmma_f32_16x16x32_bf16(false, a, false, b,
                                                       (short)0, acc[t], false, false);
    }
  }
#pragma unroll
  for (int t = 0; t < 4; ++t) {
    int col = n0 + t * 16 + l15;
    *(v8f*)(Yt + (size_t)col * M + m0 + off8) = acc[t];   // C layout: VGPR r -> M=r(+8)
  }
}

// ---------------------------------------------------------------------------
// Attention S = Q^T K per head (K-dim = 32, one wmma per 16x16 tile).
// Qh,Kh: (4,1024,32) bf16 token-major per head.  S: (4,1024,1024) f32.
// ---------------------------------------------------------------------------
__global__ void attn_scores(const __bf16* __restrict__ Qh, const __bf16* __restrict__ Kh,
                            float* __restrict__ S) {
  int wid  = (blockIdx.x * blockDim.x + threadIdx.x) >> 5;
  int lane = threadIdx.x & 31;
  int ms = wid & 15; int tmp = wid >> 4;
  int nt = tmp & 63; int h = tmp >> 6;
  int n0 = nt << 4, m0 = ms << 6;
  int l15 = lane & 15, half = lane >> 4, lo = half << 3, off8 = half << 3;

  const __bf16* qbase = Qh + ((size_t)((h << 10) + n0 + l15)) * 32;
  v16bf a = load_ab(qbase, lo);
  float* Sh = S + ((size_t)h << 20);
#pragma unroll
  for (int t = 0; t < 4; ++t) {
    const __bf16* kbase = Kh + ((size_t)((h << 10) + m0 + t * 16 + l15)) * 32;
    v16bf b = load_ab(kbase, lo);
    v8f c = {};
    c = __builtin_amdgcn_wmma_f32_16x16x32_bf16(false, a, false, b, (short)0, c, false, false);
    int col = m0 + t * 16 + l15;
#pragma unroll
    for (int r = 0; r < 8; ++r)
      Sh[(size_t)(n0 + off8 + r) * 1024 + col] = c[r];
  }
}

// Softmax over last dim (1024), with 1/sqrt(32) scale, bf16 output.
__global__ void softmax_rows(const float* __restrict__ S, __bf16* __restrict__ P) {
  __shared__ float red[256];
  int row = blockIdx.x, t = threadIdx.x;
  const float* src = S + (size_t)row * 1024;
  const float scale = 0.17677669529663687f;
  float vals[4], mx = -3.4e38f;
#pragma unroll
  for (int k = 0; k < 4; ++k) { vals[k] = src[t + (k << 8)] * scale; mx = fmaxf(mx, vals[k]); }
  red[t] = mx; __syncthreads();
  for (int s = 128; s; s >>= 1) { if (t < s) red[t] = fmaxf(red[t], red[t + s]); __syncthreads(); }
  mx = red[0]; __syncthreads();
  float e[4], sum = 0.f;
#pragma unroll
  for (int k = 0; k < 4; ++k) { e[k] = expf(vals[k] - mx); sum += e[k]; }
  red[t] = sum; __syncthreads();
  for (int s = 128; s; s >>= 1) { if (t < s) red[t] += red[t + s]; __syncthreads(); }
  float inv = 1.0f / red[0];
  __bf16* dst = P + (size_t)row * 1024;
#pragma unroll
  for (int k = 0; k < 4; ++k) dst[t + (k << 8)] = (__bf16)(e[k] * inv);
}

// O = P * V^T per head; write into Msg token-major (n,128) at channel hd*4+h.
// P: (4,1024,1024) bf16;  Vc: (4,32,1024) bf16.
__global__ void attn_out(const __bf16* __restrict__ P, const __bf16* __restrict__ Vc,
                         float* __restrict__ Msg) {
  int wid  = (blockIdx.x * blockDim.x + threadIdx.x) >> 5;
  int lane = threadIdx.x & 31;
  int nt = wid & 63; int h = wid >> 6;      // 256 waves
  int n0 = nt << 4;
  int l15 = lane & 15, half = lane >> 4, lo = half << 3, off8 = half << 3;
  v8f zero = {};
  v8f acc[2]; acc[0] = zero; acc[1] = zero;
  const __bf16* prow = P + ((size_t)(h << 10) + n0 + l15) * 1024;
  for (int k = 0; k < 1024; k += 32) {
    v16bf a = load_ab(prow + k, lo);
#pragma unroll
    for (int ct = 0; ct < 2; ++ct) {
      const __bf16* vrow = Vc + ((size_t)(h * 32 + ct * 16 + l15)) * 1024 + k;
      v16bf b = load_ab(vrow, lo);
      acc[ct] = __builtin_amdgcn_wmma_f32_16x16x32_bf16(false, a, false, b,
                                                        (short)0, acc[ct], false, false);
    }
  }
#pragma unroll
  for (int ct = 0; ct < 2; ++ct) {
    int ch = ct * 16 + l15;
#pragma unroll
    for (int r = 0; r < 8; ++r)
      Msg[(size_t)(n0 + off8 + r) * 128 + (ch << 2) + h] = acc[ct][r];
  }
}

// Final match scores: Z[n,m] = dot(m0t[n], m1t[m]) / sqrt(128); write Z and Z^T.
__global__ void match_scores_k(const __bf16* __restrict__ M0, const __bf16* __restrict__ M1,
                               float* __restrict__ Z, float* __restrict__ Zt) {
  int wid  = (blockIdx.x * blockDim.x + threadIdx.x) >> 5;
  int lane = threadIdx.x & 31;
  int ms = wid & 15; int nt = wid >> 4;    // 1024 waves
  int n0 = nt << 4, m0b = ms << 6;
  int l15 = lane & 15, half = lane >> 4, lo = half << 3, off8 = half << 3;
  const float scale = 0.08838834764831845f;
  const __bf16* arow = M0 + (size_t)(n0 + l15) * 128;
  v8f zero = {};
  v8f acc[4]; for (int t = 0; t < 4; ++t) acc[t] = zero;
  for (int k = 0; k < 128; k += 32) {
    v16bf a = load_ab(arow + k, lo);
#pragma unroll
    for (int t = 0; t < 4; ++t) {
      const __bf16* brow = M1 + (size_t)(m0b + t * 16 + l15) * 128 + k;
      v16bf b = load_ab(brow, lo);
      acc[t] = __builtin_amdgcn_wmma_f32_16x16x32_bf16(false, a, false, b,
                                                       (short)0, acc[t], false, false);
    }
  }
#pragma unroll
  for (int t = 0; t < 4; ++t) {
    int col = m0b + t * 16 + l15;
    v8f sc;
#pragma unroll
    for (int r = 0; r < 8; ++r) sc[r] = acc[t][r] * scale;
    *(v8f*)(Zt + (size_t)col * ZLD + n0 + off8) = sc;       // transposed, contiguous
#pragma unroll
    for (int r = 0; r < 8; ++r)
      Z[(size_t)(n0 + off8 + r) * ZLD + col] = sc[r];
  }
}

// ------------------------- small elementwise / VALU kernels -----------------
__global__ void f32_to_bf16_k(const float* __restrict__ x, __bf16* __restrict__ y, int n) {
  int i = blockIdx.x * blockDim.x + threadIdx.x;
  if (i < n) y[i] = (__bf16)x[i];
}

__global__ void kenc_l1(const float* __restrict__ kpts, const float* __restrict__ sc,
                        const float* __restrict__ w1, const float* __restrict__ b1,
                        float* __restrict__ y) {
  int n = blockIdx.x * blockDim.x + threadIdx.x;
  if (n >= NTOK) return;
  const float inv = 1.0f / 179.2f;          // 0.7 * max(30,256,256)
  float x0 = (kpts[n * 3 + 0] - 15.f)  * inv;
  float x1 = (kpts[n * 3 + 1] - 128.f) * inv;
  float x2 = (kpts[n * 3 + 2] - 128.f) * inv;
  float x3 = sc[n];
  for (int o = 0; o < 32; ++o) {
    const float* w = w1 + o * 4;
    y[n * 32 + o] = w[0] * x0 + w[1] * x1 + w[2] * x2 + w[3] * x3 + b1[o];
  }
}

__global__ void dense_valu(const float* __restrict__ x, const float* __restrict__ w,
                           const float* __restrict__ b, float* __restrict__ y,
                           int Cin, int Cout) {
  int n = blockIdx.x * blockDim.x + threadIdx.x;
  if (n >= NTOK) return;
  const float* xr = x + (size_t)n * Cin;
  for (int o = 0; o < Cout; ++o) {
    const float* wr = w + (size_t)o * Cin;
    float acc = b[o];
    for (int i = 0; i < Cin; ++i) acc += wr[i] * xr[i];
    y[(size_t)n * Cout + o] = acc;
  }
}

// Instance norm over tokens (per channel) + ReLU; OUTBF=1 -> bf16 out, else f32.
template <int OUTBF>
__global__ void inorm_relu(const float* __restrict__ X, float* __restrict__ Yf,
                           __bf16* __restrict__ Ybf, int C) {
  __shared__ float s1[256], s2[256];
  int c = blockIdx.x, t = threadIdx.x;
  float vals[4], sum = 0.f, sq = 0.f;
#pragma unroll
  for (int k = 0; k < 4; ++k) {
    float x = X[(size_t)(t + (k << 8)) * C + c];
    vals[k] = x; sum += x; sq += x * x;
  }
  s1[t] = sum; s2[t] = sq; __syncthreads();
  for (int s = 128; s; s >>= 1) {
    if (t < s) { s1[t] += s1[t + s]; s2[t] += s2[t + s]; }
    __syncthreads();
  }
  float mean = s1[0] * (1.f / 1024.f);
  float var  = s2[0] * (1.f / 1024.f) - mean * mean;
  float rs = rsqrtf(var + 1e-5f);
#pragma unroll
  for (int k = 0; k < 4; ++k) {
    float yv = fmaxf((vals[k] - mean) * rs, 0.f);
    if (OUTBF) Ybf[(size_t)(t + (k << 8)) * C + c] = (__bf16)yv;
    else       Yf [(size_t)(t + (k << 8)) * C + c] = yv;
  }
}

__global__ void add_desc(const float* __restrict__ desc, const float* __restrict__ kout,
                         float* __restrict__ d, __bf16* __restrict__ dbf) {
  int i = blockIdx.x * blockDim.x + threadIdx.x;
  if (i >= NTOK * 128) return;
  float v = desc[i] + kout[i];
  d[i] = v; dbf[i] = (__bf16)v;
}

__global__ void res_add_cvt(float* __restrict__ d, const float* __restrict__ dl,
                            __bf16* __restrict__ dbf) {
  int i = blockIdx.x * blockDim.x + threadIdx.x;
  if (i >= NTOK * 128) return;
  float v = d[i] + dl[i];
  d[i] = v; dbf[i] = (__bf16)v;
}

__global__ void qk_reorder(const float* __restrict__ q, __bf16* __restrict__ Qh) {
  int idx = blockIdx.x * blockDim.x + threadIdx.x;
  if (idx >= NTOK * 128) return;
  int n = idx >> 7, d = idx & 127;
  int h = d & 3, hd = d >> 2;                    // reshape (128)->(32,4)
  Qh[(((size_t)(h << 10) + n) << 5) + hd] = (__bf16)q[idx];
}

__global__ void v_reorder(const float* __restrict__ v, __bf16* __restrict__ Vc) {
  int idx = blockIdx.x * blockDim.x + threadIdx.x;
  if (idx >= NTOK * 128) return;
  int m = idx >> 7, d = idx & 127;
  int h = d & 3, hd = d >> 2;
  Vc[(((size_t)(h * 32 + hd)) << 10) + m] = (__bf16)v[idx];
}

__global__ void build_ht(const __bf16* __restrict__ xbf, const float* __restrict__ mrg,
                         __bf16* __restrict__ ht) {
  int idx = blockIdx.x * blockDim.x + threadIdx.x;
  if (idx >= NTOK * 256) return;
  int n = idx >> 8, c = idx & 255;
  ht[idx] = (c < 128) ? xbf[(n << 7) + c] : (__bf16)mrg[(n << 7) + c - 128];
}

__global__ void fill_bin(float* __restrict__ Z, float* __restrict__ Zt,
                         const float* __restrict__ alpha) {
  int j = blockIdx.x * blockDim.x + threadIdx.x;
  if (j > 1024) return;
  float a = alpha[0];
  Z [(size_t)1024 * ZLD + j] = a;  Z [(size_t)j * ZLD + 1024] = a;
  Zt[(size_t)1024 * ZLD + j] = a;  Zt[(size_t)j * ZLD + 1024] = a;
}

__global__ void zero_f32(float* __restrict__ p, int n) {
  int i = blockIdx.x * blockDim.x + threadIdx.x;
  if (i < n) p[i] = 0.f;
}

// One Sinkhorn half-step: out[i] = log_mu(i) - logsumexp_j(Zrow[i][j] + other[j])
__global__ void sinkhorn_step(const float* __restrict__ Z, const float* __restrict__ other,
                              float* __restrict__ out) {
  __shared__ float red[256];
  int i = blockIdx.x, t = threadIdx.x;
  const float* row = Z + (size_t)i * ZLD;
  float mx = -3.4e38f;
  for (int j = t; j < 1025; j += 256) mx = fmaxf(mx, row[j] + other[j]);
  red[t] = mx; __syncthreads();
  for (int s = 128; s; s >>= 1) { if (t < s) red[t] = fmaxf(red[t], red[t + s]); __syncthreads(); }
  mx = red[0]; __syncthreads();
  float sum = 0.f;
  for (int j = t; j < 1025; j += 256) sum += expf(row[j] + other[j] - mx);
  red[t] = sum; __syncthreads();
  for (int s = 128; s; s >>= 1) { if (t < s) red[t] += red[t + s]; __syncthreads(); }
  if (t == 0) {
    float lmu = (i < 1024) ? NORMC : (NORMC + LOG1024);
    out[i] = lmu - (mx + logf(red[0]));
  }
}

// Row argmax of final Zf[i,j] = Zrow[i][j] + uu[i] + vv[j] - NORMC over j<1024.
__global__ void argmax_rows(const float* __restrict__ Z, const float* __restrict__ uu,
                            const float* __restrict__ vv, float* __restrict__ mval,
                            int* __restrict__ midx) {
  __shared__ float rv[256]; __shared__ int ri[256];
  int i = blockIdx.x, t = threadIdx.x;
  const float* row = Z + (size_t)i * ZLD;
  float best = -3.4e38f; int bj = 0;
  for (int j = t; j < 1024; j += 256) {
    float val = row[j] + vv[j];
    if (val > best) { best = val; bj = j; }
  }
  rv[t] = best; ri[t] = bj; __syncthreads();
  for (int s = 128; s; s >>= 1) {
    if (t < s) {
      if (rv[t + s] > rv[t] || (rv[t + s] == rv[t] && ri[t + s] < ri[t])) {
        rv[t] = rv[t + s]; ri[t] = ri[t + s];
      }
    }
    __syncthreads();
  }
  if (t == 0) { mval[i] = rv[0] + uu[i] - NORMC; midx[i] = ri[0]; }
}

__global__ void finalize_k(const int* __restrict__ i0, const int* __restrict__ i1,
                           const float* __restrict__ m0, const float* __restrict__ m1,
                           float* __restrict__ out) {
  int n = blockIdx.x * blockDim.x + threadIdx.x;
  if (n >= NTOK) return;
  int i0n = i0[n];
  bool mut0 = (n == i1[i0n]);
  float ms0 = mut0 ? expf(m0[n]) : 0.f;
  bool v0 = mut0 && (ms0 > 0.2f);
  out[n] = v0 ? (float)i0n : -1.f;
  int i1n = i1[n];
  bool mut1 = (n == i0[i1n]);
  int tt = i1n;
  bool mut0t = (tt == i1[i0[tt]]);
  float ms0t = mut0t ? expf(m0[tt]) : 0.f;
  float ms1 = mut1 ? ms0t : 0.f;
  bool v0t = mut0t && (ms0t > 0.2f);
  bool v1 = mut1 && v0t;
  out[1024 + n] = v1 ? (float)i1n : -1.f;
  out[2048 + n] = ms0;
  out[3072 + n] = ms1;
  (void)m1;
}

__global__ void loss_k(const float* __restrict__ Z, const float* __restrict__ u,
                       const float* __restrict__ v, const int* __restrict__ am,
                       float* __restrict__ out) {
  __shared__ float red[512];
  int t = threadIdx.x;
  int x = am[2 * t], y = am[2 * t + 1];
  float zf = Z[(size_t)x * ZLD + y] + u[x] + v[y] - NORMC;
  red[t] = -logf(expf(zf) + 1e-6f);
  __syncthreads();
  for (int s = 256; s; s >>= 1) { if (t < s) red[t] += red[t + s]; __syncthreads(); }
  if (t == 0) out[0] = red[0] / 512.0f;
}

// ---------------------------------------------------------------------------
extern "C" void kernel_launch(void* const* d_in, const int* in_sizes, int n_in,
                              void* d_out, int out_size, void* d_ws, size_t ws_size,
                              hipStream_t stream) {
  (void)in_sizes; (void)n_in; (void)out_size; (void)ws_size;
  const float* desc0   = (const float*)d_in[0];
  const float* desc1   = (const float*)d_in[1];
  const float* kpts0   = (const float*)d_in[2];
  const float* kpts1   = (const float*)d_in[3];
  const float* sc0     = (const float*)d_in[4];
  const float* sc1     = (const float*)d_in[5];
  const int*   amatch  = (const int*)d_in[8];
  const float* ke_w1   = (const float*)d_in[9];
  const float* ke_b1   = (const float*)d_in[10];
  const float* ke_w2   = (const float*)d_in[11];
  const float* ke_b2   = (const float*)d_in[12];
  const float* ke_w3   = (const float*)d_in[13];
  const float* ke_b3   = (const float*)d_in[14];
  const float* proj_w  = (const float*)d_in[15];
  const float* proj_b  = (const float*)d_in[16];
  const float* merge_w = (const float*)d_in[17];
  const float* merge_b = (const float*)d_in[18];
  const float* mlp_w1  = (const float*)d_in[19];
  const float* mlp_b1  = (const float*)d_in[20];
  const float* mlp_w2  = (const float*)d_in[21];
  const float* mlp_b2  = (const float*)d_in[22];
  const float* fp_w    = (const float*)d_in[23];
  const float* fp_b    = (const float*)d_in[24];
  const float* binsc   = (const float*)d_in[25];

  // ---- bump allocator over workspace ----
  char* ws = (char*)d_ws;
  size_t off = 0;
  auto alloc = [&](size_t bytes) -> void* {
    void* p = ws + off;
    off = (off + bytes + 255) & ~(size_t)255;
    return p;
  };
  const int NE = NTOK * 128;           // 131072
  // bf16 weights
  __bf16* kew3b  = (__bf16*)alloc(128 * 64 * 2);
  __bf16* projwb = (__bf16*)alloc((size_t)18 * 3 * 16384 * 2);
  __bf16* mrgwb  = (__bf16*)alloc((size_t)18 * 16384 * 2);
  __bf16* w1b    = (__bf16*)alloc((size_t)18 * 65536 * 2);
  __bf16* w2b    = (__bf16*)alloc((size_t)18 * 32768 * 2);
  __bf16* fpwb   = (__bf16*)alloc(16384 * 2);
  // activations
  float*  d0f  = (float*)alloc(NE * 4);   __bf16* d0b = (__bf16*)alloc(NE * 2);
  float*  d1f  = (float*)alloc(NE * 4);   __bf16* d1b = (__bf16*)alloc(NE * 2);
  float*  t32  = (float*)alloc(NTOK * 32 * 4);
  float*  t32n = (float*)alloc(NTOK * 32 * 4);
  float*  t64  = (float*)alloc(NTOK * 64 * 4);
  float*  t64n = (float*)alloc(NTOK * 64 * 4);
  __bf16* t64b = (__bf16*)alloc(NTOK * 64 * 2);
  float*  kout = (float*)alloc(NE * 4);
  float*  qf   = (float*)alloc(NE * 4);
  float*  kf   = (float*)alloc(NE * 4);
  float*  vf   = (float*)alloc(NE * 4);
  __bf16* Qh   = (__bf16*)alloc(NE * 2);
  __bf16* Kh   = (__bf16*)alloc(NE * 2);
  __bf16* Vc   = (__bf16*)alloc(NE * 2);
  float*  S    = (float*)alloc((size_t)4 * 1024 * 1024 * 4);
  __bf16* P    = (__bf16*)alloc((size_t)4 * 1024 * 1024 * 2);
  float*  msgf = (float*)alloc(NE * 4);
  __bf16* msgb = (__bf16*)alloc(NE * 2);
  float*  mrgf = (float*)alloc(NE * 4);
  __bf16* htb  = (__bf16*)alloc(NTOK * 256 * 2);
  float*  h1f  = (float*)alloc(NTOK * 256 * 4);
  __bf16* h1nb = (__bf16*)alloc(NTOK * 256 * 2);
  float*  dl0  = (float*)alloc(NE * 4);
  float*  dl1  = (float*)alloc(NE * 4);
  float*  Z    = (float*)alloc((size_t)1025 * ZLD * 4);
  float*  Zt   = (float*)alloc((size_t)1025 * ZLD * 4);
  float*  uvec = (float*)alloc(ZLD * 4);
  float*  vvec = (float*)alloc(ZLD * 4);
  float*  m0v  = (float*)alloc(NTOK * 4);
  float*  m1v  = (float*)alloc(NTOK * 4);
  int*    i0b  = (int*)alloc(NTOK * 4);
  int*    i1b  = (int*)alloc(NTOK * 4);

  auto cvt = [&](const float* src, __bf16* dst, int n) {
    f32_to_bf16_k<<<(n + 255) / 256, 256, 0, stream>>>(src, dst, n);
  };
  auto gemm = [&](const __bf16* W, const __bf16* Xt, const float* bias, float* Yt,
                  int M, int K) {
    int waves = (M >> 4) * 16;          // 16 strips of 64 tokens
    gemm_wmma<<<waves / 8, 256, 0, stream>>>(W, Xt, bias, Yt, M, K);
  };

  // ---- weight conversion (once per call) ----
  cvt(ke_w3, kew3b, 128 * 64);
  cvt(proj_w, projwb, 18 * 3 * 16384);
  cvt(merge_w, mrgwb, 18 * 16384);
  cvt(mlp_w1, w1b, 18 * 65536);
  cvt(mlp_w2, w2b, 18 * 32768);
  cvt(fp_w, fpwb, 16384);

  // ---- keypoint encoder + descriptor init ----
  auto kenc = [&](const float* kp, const float* s, const float* desc,
                  float* dF, __bf16* dB) {
    kenc_l1<<<4, 256, 0, stream>>>(kp, s, ke_w1, ke_b1, t32);
    inorm_relu<0><<<32, 256, 0, stream>>>(t32, t32n, nullptr, 32);
    dense_valu<<<4, 256, 0, stream>>>(t32n, ke_w2, ke_b2, t64, 32, 64);
    inorm_relu<0><<<64, 256, 0, stream>>>(t64, t64n, nullptr, 64);
    cvt(t64n, t64b, NTOK * 64);
    gemm(kew3b, t64b, ke_b3, kout, 128, 64);
    add_desc<<<512, 256, 0, stream>>>(desc, kout, dF, dB);
  };
  kenc(kpts0, sc0, desc0, d0f, d0b);
  kenc(kpts1, sc1, desc1, d1f, d1b);

  // ---- GNN layers ----
  auto prop = [&](__bf16* xb, __bf16* sb, const __bf16* pw, const float* pb,
                  const __bf16* mw, const float* mb, const __bf16* w1,
                  const float* b1, const __bf16* w2, const float* b2, float* dl) {
    gemm(pw + 0 * 16384, xb, pb + 0,   qf, 128, 128);
    gemm(pw + 1 * 16384, sb, pb + 128, kf, 128, 128);
    gemm(pw + 2 * 16384, sb, pb + 256, vf, 128, 128);
    qk_reorder<<<512, 256, 0, stream>>>(qf, Qh);
    qk_reorder<<<512, 256, 0, stream>>>(kf, Kh);
    v_reorder<<<512, 256, 0, stream>>>(vf, Vc);
    attn_scores<<<512, 256, 0, stream>>>(Qh, Kh, S);
    softmax_rows<<<4096, 256, 0, stream>>>(S, P);
    attn_out<<<32, 256, 0, stream>>>(P, Vc, msgf);
    cvt(msgf, msgb, NE);
    gemm(mw, msgb, mb, mrgf, 128, 128);
    build_ht<<<1024, 256, 0, stream>>>(xb, mrgf, htb);
    gemm(w1, htb, b1, h1f, 256, 256);
    inorm_relu<1><<<256, 256, 0, stream>>>(h1f, nullptr, h1nb, 256);
    gemm(w2, h1nb, b2, dl, 128, 256);
  };

  for (int i = 0; i < 18; ++i) {
    __bf16* s0 = (i & 1) ? d1b : d0b;
    __bf16* s1 = (i & 1) ? d0b : d1b;
    const __bf16* pw = projwb + (size_t)i * 3 * 16384;
    const float*  pb = proj_b + (size_t)i * 3 * 128;
    const __bf16* mw = mrgwb + (size_t)i * 16384;
    const float*  mb = merge_b + (size_t)i * 128;
    const __bf16* w1 = w1b + (size_t)i * 65536;
    const float*  b1 = mlp_b1 + (size_t)i * 256;
    const __bf16* w2 = w2b + (size_t)i * 32768;
    const float*  b2 = mlp_b2 + (size_t)i * 128;
    prop(d0b, s0, pw, pb, mw, mb, w1, b1, w2, b2, dl0);
    prop(d1b, s1, pw, pb, mw, mb, w1, b1, w2, b2, dl1);
    res_add_cvt<<<512, 256, 0, stream>>>(d0f, dl0, d0b);
    res_add_cvt<<<512, 256, 0, stream>>>(d1f, dl1, d1b);
  }

  // ---- final projection + score matrix ----
  gemm(fpwb, d0b, fp_b, qf, 128, 128);
  cvt(qf, Qh, NE);
  gemm(fpwb, d1b, fp_b, kf, 128, 128);
  cvt(kf, Kh, NE);
  match_scores_k<<<128, 256, 0, stream>>>(Qh, Kh, Z, Zt);
  fill_bin<<<5, 256, 0, stream>>>(Z, Zt, binsc);

  // ---- Sinkhorn (log-OT), 100 iterations ----
  zero_f32<<<5, 256, 0, stream>>>(vvec, 1025);
  for (int it = 0; it < 100; ++it) {
    sinkhorn_step<<<1025, 256, 0, stream>>>(Z, vvec, uvec);
    sinkhorn_step<<<1025, 256, 0, stream>>>(Zt, uvec, vvec);
  }

  // ---- matching + outputs ----
  float* out = (float*)d_out;
  argmax_rows<<<1024, 256, 0, stream>>>(Z, uvec, vvec, m0v, i0b);
  argmax_rows<<<1024, 256, 0, stream>>>(Zt, vvec, uvec, m1v, i1b);
  finalize_k<<<4, 256, 0, stream>>>(i0b, i1b, m0v, m1v, out);
  loss_k<<<1, 512, 0, stream>>>(Z, uvec, vvec, amatch, out + 4096);
}
